// MoELayer_86406152061635
// MI455X (gfx1250) — matrix operand order repaired and checked by reading, not compiled, not run on hardware
//
#include <hip/hip_runtime.h>
#include <cmath>

// Problem constants (B=2, S=2048, D=1024, E=8, F=4096)
#define T_TOK 4096
#define DDIM  1024
#define NEXP  8
#define FDIM  4096
#define FCHUNK 512
#define MTILE 32            // tokens per block (2 x 16-row WMMA tiles)

typedef __attribute__((ext_vector_type(16))) __bf16       v16bf;
typedef __attribute__((ext_vector_type(8)))  float        v8f;
typedef __attribute__((ext_vector_type(8)))  unsigned int v8u;

__device__ __forceinline__ unsigned short f2bf(float f) {
  unsigned u = __float_as_uint(f);
  unsigned r = u + 0x7FFFu + ((u >> 16) & 1u);  // RNE
  return (unsigned short)(r >> 16);
}
__device__ __forceinline__ __bf16 f2bf16(float f) {
  return __builtin_bit_cast(__bf16, f2bf(f));
}
__device__ __forceinline__ float gelu_erf(float x) {
  return 0.5f * x * (1.0f + erff(x * 0.70710678118654752440f));
}
__device__ __forceinline__ v8f wmma_bf16(v16bf a, v16bf b, v8f c) {
  return __builtin_amdgcn_wmma_f32_16x16x32_bf16(false, a, false, b, (short)0, c,
                                                 false, false);
}

// ---- Weight pre-pack: fp32 (rows x N) -> bf16 B-fragment tiles --------------
// One wave packs one 32x16 (K x N) tile into 32 lanes x 16 bf16 (1KB contiguous).
// B-fragment layout: lane = n + 16*half; lane holds K = half*16 + {0..15},
// packed 2-per-dword (dword j = K pair {half*16+2j, half*16+2j+1}).
__global__ __launch_bounds__(256) void pack_w(const float* __restrict__ src,
                                              __bf16* __restrict__ dst, int N) {
  const int wave = threadIdx.x >> 5;
  const int lane = threadIdx.x & 31;
  const int half = lane >> 4;
  const long tile = (long)blockIdx.x * 8 + wave;
  const int  NT  = N >> 4;
  const long k0  = (tile / NT) * 32;           // flattened over (E*K) rows
  const int  n   = (int)(tile % NT) * 16 + (lane & 15);
  unsigned vals[8];
#pragma unroll
  for (int j = 0; j < 8; ++j) {
    float b0 = src[(k0 + half * 16 + 2 * j) * (long)N + n];
    float b1 = src[(k0 + half * 16 + 2 * j + 1) * (long)N + n];
    vals[j] = (unsigned)f2bf(b0) | ((unsigned)f2bf(b1) << 16);
  }
  v8u* out = (v8u*)(dst + tile * 512 + lane * 16);
  v8u v;
#pragma unroll
  for (int j = 0; j < 8; ++j) v[j] = vals[j];
  *out = v;
}

// ---------------- Router: top-1 over E=8, one wave per token ----------------
__global__ __launch_bounds__(256) void moe_router(const float* __restrict__ x,
                                                  const float* __restrict__ gw,
                                                  int* __restrict__ eid,
                                                  int* __restrict__ counts) {
  int token = blockIdx.x * 8 + (threadIdx.x >> 5);
  int lane  = threadIdx.x & 31;
  if (token >= T_TOK) return;
  const float* xr = x + (long)token * DDIM;
  float acc[NEXP];
#pragma unroll
  for (int e = 0; e < NEXP; ++e) acc[e] = 0.0f;
  for (int d = lane; d < DDIM; d += 32) {
    float xv = xr[d];
#pragma unroll
    for (int e = 0; e < NEXP; ++e) acc[e] += xv * gw[e * DDIM + d];
  }
#pragma unroll
  for (int off = 16; off > 0; off >>= 1) {
#pragma unroll
    for (int e = 0; e < NEXP; ++e) acc[e] += __shfl_xor(acc[e], off, 32);
  }
  if (lane == 0) {
    int best = 0;
    float bv = acc[0];
#pragma unroll
    for (int e = 1; e < NEXP; ++e)
      if (acc[e] > bv) { bv = acc[e]; best = e; }  // first-max == jnp.argmax
    eid[token] = best;
    atomicAdd(&counts[best], 1);
  }
}

__global__ void moe_scan(const int* __restrict__ counts, int* __restrict__ offsets) {
  if (threadIdx.x == 0) {
    int s = 0;
    for (int e = 0; e < NEXP; ++e) { offsets[e] = s; s += counts[e]; }
  }
}

__global__ __launch_bounds__(256) void moe_scatter(const int* __restrict__ eid,
                                                   const int* __restrict__ offsets,
                                                   int* __restrict__ fill,
                                                   int* __restrict__ bucket) {
  int t = blockIdx.x * 256 + threadIdx.x;
  if (t >= T_TOK) return;
  int e = eid[t];
  int pos = atomicAdd(&fill[e], 1);
  bucket[offsets[e] + pos] = t;
}

// A-fragment slot for element (m, k) of a 16x32 bf16 A tile:
//   lane = m + 16*(k>>4), idx = k&15
__device__ __forceinline__ int a_slot(int m, int kk) {
  int dl  = m + ((kk >> 4) << 4);
  int idx = kk & 15;
  return dl * 16 + idx;
}

// ------- Grouped FFN: one block = one 32-token tile of one expert (512 thr) -------
// Each wave processes BOTH 16-row M-tiles against its B tiles: every B fragment
// fetched from L2/L0 feeds 2 WMMAs (2x arithmetic intensity on the B stream).
__global__ __launch_bounds__(512) void moe_ffn(const float* __restrict__ x,
                                               const __bf16* __restrict__ pw1,
                                               const __bf16* __restrict__ pw2,
                                               const int* __restrict__ counts,
                                               const int* __restrict__ offsets,
                                               const int* __restrict__ bucket,
                                               float* __restrict__ out) {
  const int e    = blockIdx.x >> 7;   // 8 experts x 128 tile slots
  const int tile = blockIdx.x & 127;
  const int cnt  = counts[e];
  if (tile * MTILE >= cnt) return;
  const int base = offsets[e] + tile * MTILE;

  __shared__ int tok[MTILE];
  // A-fragment-layout LDS: per mtile, [ktile][lane*16 + idx]
  __shared__ __align__(64) __bf16 Xs[2 * 32 * 512];  // 32 tokens x 1024  (64KB)
  __shared__ __align__(64) __bf16 Hs[2 * 16 * 512];  // 32 tokens x 512   (32KB)

  const int tid  = threadIdx.x;
  const int wave = tid >> 5;          // 0..15
  const int lane = tid & 31;
  const int half = lane >> 4;
  const int lrow = lane & 15;

  if (tid < MTILE) tok[tid] = (tile * MTILE + tid < cnt) ? bucket[base + tid] : -1;
  __syncthreads();

  // Stage X tile into LDS in A-fragment layout (coalesced global reads)
  for (int i = tid; i < MTILE * DDIM; i += 512) {
    int m32 = i >> 10, c = i & (DDIM - 1);
    int t = tok[m32];
    float v = (t >= 0) ? x[(long)t * DDIM + c] : 0.0f;
    int mt = m32 >> 4, m = m32 & 15;
    Xs[mt * 16384 + ((c >> 5) << 9) + a_slot(m, c & 31)] = f2bf16(v);
  }
  __syncthreads();

  const __bf16* PW1 = pw1 + (long)e * DDIM * FDIM;  // (K/32=32, N/16=256) tiles
  const __bf16* PW2 = pw2 + (long)e * FDIM * DDIM;  // (K/32=128, N/16=64) tiles

  v8f oacc[2][4];  // wave owns out N-tiles {wave*4..wave*4+3} for both mtiles
#pragma unroll
  for (int mt = 0; mt < 2; ++mt)
#pragma unroll
    for (int ot = 0; ot < 4; ++ot) oacc[mt][ot] = (v8f){};

  for (int fc = 0; fc < FDIM; fc += FCHUNK) {
    // ---- stage 1: h = gelu(X @ W1[:, fc:fc+512]); wave owns 2 N-tiles x 2 M ----
    v8f hacc[2][2];
#pragma unroll
    for (int mt = 0; mt < 2; ++mt)
#pragma unroll
      for (int nt = 0; nt < 2; ++nt) hacc[mt][nt] = (v8f){};
    const int ntg0 = (fc >> 4) + wave * 2;  // global N-tile index of nt=0
    for (int k0 = 0; k0 < DDIM; k0 += 32) {
      const int kt = k0 >> 5;
      v16bf a0 = __builtin_bit_cast(
          v16bf, *(const v8u*)(Xs + (kt << 9) + lane * 16));
      v16bf a1 = __builtin_bit_cast(
          v16bf, *(const v8u*)(Xs + 16384 + (kt << 9) + lane * 16));
#pragma unroll
      for (int nt = 0; nt < 2; ++nt) {
        const v8u* bp =
            (const v8u*)(PW1 + ((long)kt * 256 + (ntg0 + nt)) * 512 + lane * 16);
        v16bf b = __builtin_bit_cast(v16bf, *bp);
        hacc[0][nt] = wmma_bf16(a0, b, hacc[0][nt]);
        hacc[1][nt] = wmma_bf16(a1, b, hacc[1][nt]);
      }
    }
    // GELU + store h into LDS in A-fragment layout for stage 2
#pragma unroll
    for (int mt = 0; mt < 2; ++mt) {
#pragma unroll
      for (int nt = 0; nt < 2; ++nt) {
        int f = (wave * 2 + nt) * 16 + lrow;  // column within chunk [0,512)
#pragma unroll
        for (int r = 0; r < 8; ++r) {
          int m = 8 * half + r;  // C-layout row
          Hs[mt * 8192 + ((f >> 5) << 9) + a_slot(m, f & 31)] =
              f2bf16(gelu_erf(hacc[mt][nt][r]));
        }
      }
    }
    __syncthreads();

    // ---- stage 2: out += h @ W2[fc:fc+512, :] ----
    for (int k0 = 0; k0 < FCHUNK; k0 += 32) {
      v16bf a0 = __builtin_bit_cast(
          v16bf, *(const v8u*)(Hs + ((k0 >> 5) << 9) + lane * 16));
      v16bf a1 = __builtin_bit_cast(
          v16bf, *(const v8u*)(Hs + 8192 + ((k0 >> 5) << 9) + lane * 16));
      const long ktg = (long)((fc + k0) >> 5);
#pragma unroll
      for (int ot = 0; ot < 4; ++ot) {
        const v8u* bp =
            (const v8u*)(PW2 + (ktg * 64 + (wave * 4 + ot)) * 512 + lane * 16);
        v16bf b = __builtin_bit_cast(v16bf, *bp);
        oacc[0][ot] = wmma_bf16(a0, b, oacc[0][ot]);
        oacc[1][ot] = wmma_bf16(a1, b, oacc[1][ot]);
      }
    }
    __syncthreads();  // before next chunk overwrites Hs
  }

  // Scatter fp32 output rows
#pragma unroll
  for (int mt = 0; mt < 2; ++mt) {
#pragma unroll
    for (int ot = 0; ot < 4; ++ot) {
      int ncol = (wave * 4 + ot) * 16 + lrow;
#pragma unroll
      for (int r = 0; r < 8; ++r) {
        int m = 8 * half + r;
        int t = tok[mt * 16 + m];
        if (t >= 0) out[(long)t * DDIM + ncol] = oacc[mt][ot][r];
      }
    }
  }
}

extern "C" void kernel_launch(void* const* d_in, const int* in_sizes, int n_in,
                              void* d_out, int out_size, void* d_ws, size_t ws_size,
                              hipStream_t stream) {
  (void)in_sizes; (void)n_in; (void)out_size; (void)ws_size;
  const float* x  = (const float*)d_in[0];  // (B,S,D)
  const float* gw = (const float*)d_in[1];  // (E,D)
  const float* w1 = (const float*)d_in[2];  // (E,D,F)
  const float* w2 = (const float*)d_in[3];  // (E,F,D)
  float* out = (float*)d_out;

  // Workspace: ints | (64KB aligned) pw1 bf16 | pw2 bf16
  char* ws = (char*)d_ws;
  int* eid     = (int*)ws;
  int* counts  = eid + T_TOK;
  int* fill    = counts + NEXP;
  int* offsets = fill + NEXP;
  int* bucket  = offsets + NEXP;
  __bf16* pw1 = (__bf16*)(ws + 65536);
  __bf16* pw2 = pw1 + (long)NEXP * DDIM * FDIM;

  hipMemsetAsync(counts, 0, 2 * NEXP * sizeof(int), stream);  // counts + fill

  // Pack weights into bf16 B-fragment tiles (one wave per 32x16 tile)
  {
    long tiles1 = ((long)NEXP * DDIM / 32) * (FDIM / 16);  // 65536
    long tiles2 = ((long)NEXP * FDIM / 32) * (DDIM / 16);  // 65536
    pack_w<<<(int)(tiles1 / 8), 256, 0, stream>>>(w1, pw1, FDIM);
    pack_w<<<(int)(tiles2 / 8), 256, 0, stream>>>(w2, pw2, DDIM);
  }

  moe_router<<<T_TOK / 8, 256, 0, stream>>>(x, gw, eid, counts);
  moe_scan<<<1, 32, 0, stream>>>(counts, offsets);
  moe_scatter<<<T_TOK / 256, 256, 0, stream>>>(eid, offsets, fill, bucket);
  moe_ffn<<<NEXP * 128, 512, 0, stream>>>(x, pw1, pw2, counts, offsets, bucket, out);
}
